// HGT_built_in_sample_69965017252746
// MI455X (gfx1250) — matrix operand order repaired and checked by reading, not compiled, not run on hardware
//
#include <hip/hip_runtime.h>
#include <hip/hip_bf16.h>
#include <math.h>

#define N_NODES 50000
#define N_EDGES 400000
#define NINP    256
#define HID     128
#define NHEAD   4
#define DHEAD   32
#define TNT     4
#define TET     6
#define NLAYER  2
#define NOUT    16

typedef __attribute__((ext_vector_type(16))) __bf16 v16bf;
typedef __attribute__((ext_vector_type(8)))  float  v8f;

struct BF16x16raw { uint4 lo, hi; };   // 32 bytes == v16bf

static __device__ __forceinline__ v8f wmma_bf16(v16bf a, v16bf b, v8f c) {
  // D(16x16,f32) = A(16x32,bf16) x B(32x16,bf16) + C
  return __builtin_amdgcn_wmma_f32_16x16x32_bf16(false, a, false, b, (short)0, c, false, false);
}

// A tile (16x32) from fp32 row-major source: 4x b128 loads + pk converts.
// Layout (ISA 7.12.2, 16-bit A): lane m=lane&15, elems 0..7 -> K=k0+8*hf+e,
// elems 8..15 -> K=k0+16+8*hf+(e-8)  (two contiguous runs of 8).
static __device__ __forceinline__ v16bf load_A_f32(const float* __restrict__ base, int ld,
                                                   int k0, int lane) {
  const int hf = lane >> 4, m = lane & 15;
  const float* r = base + (size_t)m * ld + k0 + 8 * hf;
  float4 f0 = *(const float4*)(r);
  float4 f1 = *(const float4*)(r + 4);
  float4 f2 = *(const float4*)(r + 16);
  float4 f3 = *(const float4*)(r + 20);
  v16bf a;
  a[0] = (__bf16)f0.x;  a[1] = (__bf16)f0.y;  a[2] = (__bf16)f0.z;  a[3] = (__bf16)f0.w;
  a[4] = (__bf16)f1.x;  a[5] = (__bf16)f1.y;  a[6] = (__bf16)f1.z;  a[7] = (__bf16)f1.w;
  a[8] = (__bf16)f2.x;  a[9] = (__bf16)f2.y;  a[10] = (__bf16)f2.z; a[11] = (__bf16)f2.w;
  a[12] = (__bf16)f3.x; a[13] = (__bf16)f3.y; a[14] = (__bf16)f3.z; a[15] = (__bf16)f3.w;
  return a;
}

// A tile from bf16 row-major mirror: 2x b128 loads, no converts.
static __device__ __forceinline__ v16bf load_A_bf16(const __bf16* __restrict__ base, int ld,
                                                    int k0, int lane) {
  const int hf = lane >> 4, m = lane & 15;
  const __bf16* r = base + (size_t)m * ld + k0 + 8 * hf;
  BF16x16raw u;
  u.lo = *(const uint4*)(r);
  u.hi = *(const uint4*)(r + 16);
  return __builtin_bit_cast(v16bf, u);
}

// B tile (32x16) from bf16 TRANSPOSED weights wT[col][K]: lane n=lane&15 reads
// K runs k0+8*hf+0..7 and k0+16+8*hf+0..7 -> 2x b128 loads, no converts.
static __device__ __forceinline__ v16bf load_B_T(const __bf16* __restrict__ wT, int ldK,
                                                 int k0, int col0, int lane) {
  const int hf = lane >> 4, n = lane & 15;
  const __bf16* r = wT + (size_t)(col0 + n) * ldK + k0 + 8 * hf;
  BF16x16raw u;
  u.lo = *(const uint4*)(r);
  u.hi = *(const uint4*)(r + 16);
  return __builtin_bit_cast(v16bf, u);
}

// order-preserving float<->uint keys for atomicMax-based segment max
static __device__ __forceinline__ unsigned enc_ord(float f) {
  unsigned b = __float_as_uint(f);
  return (b >> 31) ? ~b : (b ^ 0x80000000u);
}
static __device__ __forceinline__ float dec_ord(unsigned k) {
  unsigned b = (k >> 31) ? (k ^ 0x80000000u) : ~k;
  return __uint_as_float(b);
}

// ---------------- prepack: dst[mat][c][k] = (bf16) src[mat][k][c] ----------------
__global__ void prepack_wT_kernel(const float* __restrict__ src, __bf16* __restrict__ dst,
                                  int K, int C) {
  const float* s = src + (size_t)blockIdx.y * K * C;
  __bf16* d = dst + (size_t)blockIdx.y * K * C;
  int i = blockIdx.x * blockDim.x + threadIdx.x;
  if (i >= K * C) return;
  int k = i / C, c = i - k * C;
  d[(size_t)c * K + k] = (__bf16)s[i];
}

// ---------------- adapt: h = GELU(x @ adapt_W[ntype] + adapt_b[ntype]) ----------------
__global__ void __launch_bounds__(256) adapt_gelu_kernel(
    const float* __restrict__ x, const __bf16* __restrict__ WT,
    const float* __restrict__ bias, const int* __restrict__ ntype,
    float* __restrict__ h, __bf16* __restrict__ hbf) {
  const int lane = threadIdx.x & 31;
  const int wv   = threadIdx.x >> 5;   // 8 waves = 8 column tiles of 16 (HID=128)
  const int row0 = blockIdx.x * 16;
  const int col0 = wv * 16;
  const float* xrow = x + (size_t)(row0 + (lane & 15)) * NINP;
  v8f acc0 = {}, acc1 = {}, acc2 = {}, acc3 = {};
  for (int k0 = 0; k0 < NINP; k0 += 32) {
    if (k0 + 32 < NINP) {                       // gfx1250 global_prefetch of next A slab
      __builtin_prefetch(xrow + k0 + 32, 0, 3);
      __builtin_prefetch(xrow + k0 + 48, 0, 3);
    }
    v16bf a = load_A_f32(x + (size_t)row0 * NINP, NINP, k0, lane);
    acc0 = wmma_bf16(a, load_B_T(WT + 0ull * NINP * HID, NINP, k0, col0, lane), acc0);
    acc1 = wmma_bf16(a, load_B_T(WT + 1ull * NINP * HID, NINP, k0, col0, lane), acc1);
    acc2 = wmma_bf16(a, load_B_T(WT + 2ull * NINP * HID, NINP, k0, col0, lane), acc2);
    acc3 = wmma_bf16(a, load_B_T(WT + 3ull * NINP * HID, NINP, k0, col0, lane), acc3);
  }
  const int hf = lane >> 4, n = lane & 15;
#pragma unroll
  for (int r = 0; r < 8; ++r) {
    int row = row0 + r + 8 * hf;
    int t = ntype[row];
    float v = (t == 0) ? acc0[r] : (t == 1) ? acc1[r] : (t == 2) ? acc2[r] : acc3[r];
    v += bias[t * HID + col0 + n];
    v = 0.5f * v * (1.0f + erff(v * 0.70710678118654752f));   // exact GELU
    h[(size_t)row * HID + col0 + n] = v;
    hbf[(size_t)row * HID + col0 + n] = (__bf16)v;
  }
}

// ---------------- typed linear [N,128] @ W[ntype] (K/Q/V), bf16 input mirror ----------------
__global__ void __launch_bounds__(256) typed_linear128_kernel(
    const __bf16* __restrict__ inbf, const __bf16* __restrict__ WT,
    const int* __restrict__ ntype, float* __restrict__ out) {
  const int lane = threadIdx.x & 31;
  const int wv   = threadIdx.x >> 5;
  const int row0 = blockIdx.x * 16;
  const int col0 = wv * 16;
  v8f acc0 = {}, acc1 = {}, acc2 = {}, acc3 = {};
#pragma unroll
  for (int k0 = 0; k0 < HID; k0 += 32) {
    v16bf a = load_A_bf16(inbf + (size_t)row0 * HID, HID, k0, lane);
    acc0 = wmma_bf16(a, load_B_T(WT + 0ull * HID * HID, HID, k0, col0, lane), acc0);
    acc1 = wmma_bf16(a, load_B_T(WT + 1ull * HID * HID, HID, k0, col0, lane), acc1);
    acc2 = wmma_bf16(a, load_B_T(WT + 2ull * HID * HID, HID, k0, col0, lane), acc2);
    acc3 = wmma_bf16(a, load_B_T(WT + 3ull * HID * HID, HID, k0, col0, lane), acc3);
  }
  const int hf = lane >> 4, n = lane & 15;
#pragma unroll
  for (int r = 0; r < 8; ++r) {
    int row = row0 + r + 8 * hf;
    int t = ntype[row];
    float v = (t == 0) ? acc0[r] : (t == 1) ? acc1[r] : (t == 2) ? acc2[r] : acc3[r];
    out[(size_t)row * HID + col0 + n] = v;
  }
}

// ---------------- init segment-softmax state ----------------
__global__ void init_seg_kernel(float* __restrict__ agg, unsigned* __restrict__ keys,
                                float* __restrict__ den) {
  int i = blockIdx.x * blockDim.x + threadIdx.x;
  if (i < N_NODES * HID) agg[i] = 0.0f;
  if (i < N_NODES * NHEAD) { keys[i] = 0x00800000u; /* enc(-FLT_MAX) */ den[i] = 0.0f; }
}

// ---------------- per-(edge,head) attention score + segment max ----------------
__global__ void __launch_bounds__(256) edge_score_kernel(
    const float* __restrict__ kbuf, const float* __restrict__ qbuf,
    const int* __restrict__ etype, const int* __restrict__ srcv,
    const int* __restrict__ dstv, const float* __restrict__ Wa_l,
    const float* __restrict__ pri_l, float* __restrict__ a_out,
    unsigned* __restrict__ keys) {
  const int gw = (int)((blockIdx.x * blockDim.x + threadIdx.x) >> 5);
  const int lane = threadIdx.x & 31;
  const int e = gw >> 2, hd = gw & 3;
  if (e >= N_EDGES) return;
  const int t = etype[e], s = srcv[e], d = dstv[e];
  const float kv = kbuf[(size_t)s * HID + hd * DHEAD + lane];
  const float qv = qbuf[(size_t)d * HID + hd * DHEAD + lane];
  const float* __restrict__ WA = Wa_l + (size_t)(hd * TET + t) * DHEAD * DHEAD;
  float kw = 0.0f;   // kw[lane] = sum_d k[d] * Wa[d][lane]
#pragma unroll
  for (int dd = 0; dd < DHEAD; ++dd)
    kw = fmaf(__shfl(kv, dd, 32), WA[dd * DHEAD + lane], kw);
  float sc = kw * qv;
#pragma unroll
  for (int m = 16; m; m >>= 1) sc += __shfl_xor(sc, m, 32);
  sc = sc * pri_l[hd * TET + t] * 0.17677669529663687f;   // 1/sqrt(32)
  if (lane == 0) {
    a_out[e * NHEAD + hd] = sc;
    atomicMax(&keys[d * NHEAD + hd], enc_ord(sc));
  }
}

// ---------------- exp(a - segmax) and segment denominator ----------------
__global__ void edge_exp_kernel(const float* __restrict__ a_out,
                                const unsigned* __restrict__ keys,
                                const int* __restrict__ dstv,
                                float* __restrict__ ex_out, float* __restrict__ den) {
  int idx = blockIdx.x * blockDim.x + threadIdx.x;
  if (idx >= N_EDGES * NHEAD) return;
  int e = idx >> 2, hd = idx & 3;
  int d = dstv[e];
  float amax = dec_ord(keys[d * NHEAD + hd]);
  float ex = __expf(a_out[idx] - amax);
  ex_out[idx] = ex;
  atomicAdd(&den[d * NHEAD + hd], ex);
}

// ---------------- aggregate: recompute m = v_e @ Wm[t] and atomic scatter ----------------
__global__ void __launch_bounds__(256) edge_agg_kernel(
    const float* __restrict__ vbuf, const int* __restrict__ etype,
    const int* __restrict__ srcv, const int* __restrict__ dstv,
    const float* __restrict__ Wm_l, const float* __restrict__ ex_out,
    const float* __restrict__ den, float* __restrict__ agg) {
  const int gw = (int)((blockIdx.x * blockDim.x + threadIdx.x) >> 5);
  const int lane = threadIdx.x & 31;
  const int e = gw >> 2, hd = gw & 3;
  if (e >= N_EDGES) return;
  const int t = etype[e], s = srcv[e], d = dstv[e];
  const float vv = vbuf[(size_t)s * HID + hd * DHEAD + lane];
  const float* __restrict__ WM = Wm_l + (size_t)(hd * TET + t) * DHEAD * DHEAD;
  float mw = 0.0f;
#pragma unroll
  for (int dd = 0; dd < DHEAD; ++dd)
    mw = fmaf(__shfl(vv, dd, 32), WM[dd * DHEAD + lane], mw);
  const float attn = ex_out[e * NHEAD + hd] / den[d * NHEAD + hd];
  atomicAdd(&agg[(size_t)d * HID + hd * DHEAD + lane], mw * attn);
}

// ---------------- Wla typed linear + skip mix + LayerNorm ----------------
__global__ void __launch_bounds__(256) wla_skip_ln_kernel(
    const float* __restrict__ agg, const __bf16* __restrict__ WlaT_l,
    const float* __restrict__ hprev, const int* __restrict__ ntype,
    const float* __restrict__ skip_l, const float* __restrict__ gamma_l,
    const float* __restrict__ beta_l, float* __restrict__ hout,
    __bf16* __restrict__ hbf) {
  __shared__ float lds[16 * HID];
  const int lane = threadIdx.x & 31;
  const int wv   = threadIdx.x >> 5;
  const int row0 = blockIdx.x * 16;
  const int col0 = wv * 16;
  v8f acc0 = {}, acc1 = {}, acc2 = {}, acc3 = {};
#pragma unroll
  for (int k0 = 0; k0 < HID; k0 += 32) {
    v16bf a = load_A_f32(agg + (size_t)row0 * HID, HID, k0, lane);
    acc0 = wmma_bf16(a, load_B_T(WlaT_l + 0ull * HID * HID, HID, k0, col0, lane), acc0);
    acc1 = wmma_bf16(a, load_B_T(WlaT_l + 1ull * HID * HID, HID, k0, col0, lane), acc1);
    acc2 = wmma_bf16(a, load_B_T(WlaT_l + 2ull * HID * HID, HID, k0, col0, lane), acc2);
    acc3 = wmma_bf16(a, load_B_T(WlaT_l + 3ull * HID * HID, HID, k0, col0, lane), acc3);
  }
  const int hf = lane >> 4, n = lane & 15;
#pragma unroll
  for (int r = 0; r < 8; ++r) {
    int rl = r + 8 * hf;
    int row = row0 + rl;
    int t = ntype[row];
    float ha = (t == 0) ? acc0[r] : (t == 1) ? acc1[r] : (t == 2) ? acc2[r] : acc3[r];
    float alpha = 1.0f / (1.0f + __expf(-skip_l[t]));
    float ov = ha * alpha + hprev[(size_t)row * HID + col0 + n] * (1.0f - alpha);
    lds[rl * HID + col0 + n] = ov;
  }
  __syncthreads();
  // LayerNorm over 128 cols; 16 threads per row, 8 elems each (stride 16)
  const int rr = threadIdx.x >> 4, g = threadIdx.x & 15;
  float s = 0.0f;
#pragma unroll
  for (int j = 0; j < 8; ++j) s += lds[rr * HID + g + 16 * j];
#pragma unroll
  for (int m = 8; m; m >>= 1) s += __shfl_xor(s, m, 32);
  const float mu = s * (1.0f / HID);
  float vs = 0.0f;
#pragma unroll
  for (int j = 0; j < 8; ++j) {
    float dv = lds[rr * HID + g + 16 * j] - mu;
    vs = fmaf(dv, dv, vs);
  }
#pragma unroll
  for (int m = 8; m; m >>= 1) vs += __shfl_xor(vs, m, 32);
  const float rstd = rsqrtf(vs * (1.0f / HID) + 1e-5f);
#pragma unroll
  for (int j = 0; j < 8; ++j) {
    int c = g + 16 * j;
    float xv = (lds[rr * HID + c] - mu) * rstd;
    float o = xv * gamma_l[c] + beta_l[c];
    hout[(size_t)(row0 + rr) * HID + c] = o;
    hbf[(size_t)(row0 + rr) * HID + c] = (__bf16)o;
  }
}

// ---------------- output projection h @ out_W + out_b ----------------
__global__ void __launch_bounds__(256) out_proj_kernel(
    const __bf16* __restrict__ hbf, const __bf16* __restrict__ WT,
    const float* __restrict__ bias, float* __restrict__ out) {
  const int gw = (int)((blockIdx.x * blockDim.x + threadIdx.x) >> 5);
  const int lane = threadIdx.x & 31;
  if (gw >= N_NODES / 16) return;           // wave-uniform: EXEC stays all-ones for WMMA
  const int row0 = gw * 16;
  v8f acc = {};
#pragma unroll
  for (int k0 = 0; k0 < HID; k0 += 32) {
    v16bf a = load_A_bf16(hbf + (size_t)row0 * HID, HID, k0, lane);
    v16bf b = load_B_T(WT, HID, k0, 0, lane);    // out_WT[16][128]
    acc = wmma_bf16(a, b, acc);
  }
  const int hf = lane >> 4, n = lane & 15;
#pragma unroll
  for (int r = 0; r < 8; ++r) {
    int row = row0 + r + 8 * hf;
    out[(size_t)row * NOUT + n] = acc[r] + bias[n];
  }
}

extern "C" void kernel_launch(void* const* d_in, const int* in_sizes, int n_in,
                              void* d_out, int out_size, void* d_ws, size_t ws_size,
                              hipStream_t stream) {
  const float* x       = (const float*)d_in[0];
  const float* adapt_W = (const float*)d_in[1];
  const float* adapt_b = (const float*)d_in[2];
  const float* Wk      = (const float*)d_in[3];
  const float* Wq      = (const float*)d_in[4];
  const float* Wv      = (const float*)d_in[5];
  const float* pri     = (const float*)d_in[6];
  const float* Wa      = (const float*)d_in[7];
  const float* Wm      = (const float*)d_in[8];
  const float* Wla     = (const float*)d_in[9];
  const float* skip    = (const float*)d_in[10];
  const float* gamma   = (const float*)d_in[11];
  const float* beta    = (const float*)d_in[12];
  const float* out_W   = (const float*)d_in[13];
  const float* out_b   = (const float*)d_in[14];
  const int*   ntype   = (const int*)d_in[15];
  const int*   etype   = (const int*)d_in[16];
  const int*   srcv    = (const int*)d_in[17];
  const int*   dstv    = (const int*)d_in[18];

  float* ws = (float*)d_ws;
  size_t off = 0;
  float* h0   = ws + off; off += (size_t)N_NODES * HID;
  float* h1   = ws + off; off += (size_t)N_NODES * HID;
  float* kb   = ws + off; off += (size_t)N_NODES * HID;
  float* qb   = ws + off; off += (size_t)N_NODES * HID;
  float* vb   = ws + off; off += (size_t)N_NODES * HID;
  float* agg  = ws + off; off += (size_t)N_NODES * HID;
  float* a_sc = ws + off; off += (size_t)N_EDGES * NHEAD;
  float* exb  = ws + off; off += (size_t)N_EDGES * NHEAD;
  unsigned* keys = (unsigned*)(ws + off); off += (size_t)N_NODES * NHEAD;
  float* den  = ws + off; off += (size_t)N_NODES * NHEAD;
  __bf16* hbf = (__bf16*)(ws + off); off += (size_t)N_NODES * HID / 2;
  __bf16* adapt_WT = (__bf16*)(ws + off); off += (size_t)TNT * NINP * HID / 2;
  __bf16* WkT  = (__bf16*)(ws + off); off += (size_t)NLAYER * TNT * HID * HID / 2;
  __bf16* WqT  = (__bf16*)(ws + off); off += (size_t)NLAYER * TNT * HID * HID / 2;
  __bf16* WvT  = (__bf16*)(ws + off); off += (size_t)NLAYER * TNT * HID * HID / 2;
  __bf16* WlaT = (__bf16*)(ws + off); off += (size_t)NLAYER * TNT * HID * HID / 2;
  __bf16* out_WT = (__bf16*)(ws + off); off += (size_t)HID * NOUT / 2;

  const dim3 blk(256);
  const int ntiles = N_NODES / 16;                    // 3125, exact
  const int eblocks = (N_EDGES * NHEAD) / 8;          // 200000, exact (8 waves/block)

  // ---- prepack weights (bf16, transposed) ----
  prepack_wT_kernel<<<dim3((NINP * HID + 255) / 256, TNT), blk, 0, stream>>>(adapt_W, adapt_WT, NINP, HID);
  prepack_wT_kernel<<<dim3((HID * HID + 255) / 256, NLAYER * TNT), blk, 0, stream>>>(Wk, WkT, HID, HID);
  prepack_wT_kernel<<<dim3((HID * HID + 255) / 256, NLAYER * TNT), blk, 0, stream>>>(Wq, WqT, HID, HID);
  prepack_wT_kernel<<<dim3((HID * HID + 255) / 256, NLAYER * TNT), blk, 0, stream>>>(Wv, WvT, HID, HID);
  prepack_wT_kernel<<<dim3((HID * HID + 255) / 256, NLAYER * TNT), blk, 0, stream>>>(Wla, WlaT, HID, HID);
  prepack_wT_kernel<<<dim3((HID * NOUT + 255) / 256, 1), blk, 0, stream>>>(out_W, out_WT, HID, NOUT);

  adapt_gelu_kernel<<<ntiles, blk, 0, stream>>>(x, adapt_WT, adapt_b, ntype, h0, hbf);

  float* hcur = h0;
  float* hnxt = h1;
  for (int l = 0; l < NLAYER; ++l) {
    const __bf16* WkT_l  = WkT  + (size_t)l * TNT * HID * HID;
    const __bf16* WqT_l  = WqT  + (size_t)l * TNT * HID * HID;
    const __bf16* WvT_l  = WvT  + (size_t)l * TNT * HID * HID;
    const __bf16* WlaT_l = WlaT + (size_t)l * TNT * HID * HID;
    const float* Wa_l  = Wa  + (size_t)l * NHEAD * TET * DHEAD * DHEAD;
    const float* Wm_l  = Wm  + (size_t)l * NHEAD * TET * DHEAD * DHEAD;
    const float* pri_l = pri + (size_t)l * NHEAD * TET;

    typed_linear128_kernel<<<ntiles, blk, 0, stream>>>(hbf, WkT_l, ntype, kb);
    typed_linear128_kernel<<<ntiles, blk, 0, stream>>>(hbf, WqT_l, ntype, qb);
    typed_linear128_kernel<<<ntiles, blk, 0, stream>>>(hbf, WvT_l, ntype, vb);

    init_seg_kernel<<<(N_NODES * HID + 255) / 256, blk, 0, stream>>>(agg, keys, den);
    edge_score_kernel<<<eblocks, blk, 0, stream>>>(kb, qb, etype, srcv, dstv,
                                                   Wa_l, pri_l, a_sc, keys);
    edge_exp_kernel<<<(N_EDGES * NHEAD + 255) / 256, blk, 0, stream>>>(a_sc, keys, dstv, exb, den);
    edge_agg_kernel<<<eblocks, blk, 0, stream>>>(vb, etype, srcv, dstv, Wm_l, exb, den, agg);

    // writes next h (fp32) AND refreshes the bf16 mirror used by next layer's K/Q/V
    wla_skip_ln_kernel<<<ntiles, blk, 0, stream>>>(agg, WlaT_l, hcur, ntype,
                                                   skip + (size_t)l * TNT,
                                                   gamma + (size_t)l * HID,
                                                   beta + (size_t)l * HID, hnxt, hbf);
    float* tmp = hcur; hcur = hnxt; hnxt = tmp;
  }

  out_proj_kernel<<<(ntiles + 7) / 8, blk, 0, stream>>>(hbf, out_WT, out_b, (float*)d_out);
}
